// PolicyModel_43559558316645
// MI455X (gfx1250) — matrix operand order repaired and checked by reading, not compiled, not run on hardware
//
#include <hip/hip_runtime.h>
#include <hip/hip_bf16.h>
#include <stdint.h>
#include <stddef.h>

// ---------------------------------------------------------------------------
// Problem constants (fixed by the reference implementation)
// ---------------------------------------------------------------------------
#define HDIM   1024
#define GATES  4096          // 4*HDIM
#define NOPS   32000
#define NLAYER 128
#define NLSTM  2

typedef __attribute__((ext_vector_type(16))) __bf16 v16bf;
typedef __attribute__((ext_vector_type(8)))  __bf16 v8bf;
typedef __attribute__((ext_vector_type(8)))  float  v8f;

// ---------------------------------------------------------------------------
// Wave-level 16-row matvec strip using v_wmma_f32_16x16x32_bf16.
//
// A tile layout (16-bit A 16x32, ISA 7.12.2): lane m (0-15) = row M=m,
//   half h: VGPR0..3 hold K = h*8..h*8+7, VGPR4..7 hold K = 16+h*8..23+h*8.
// B (32x16): lane n (0-15) = column n with K=0..7,16..23; lane n+16 = column n
//   with K=8..15,24..31.  We BROADCAST the x-chunk into every column (the load
//   address depends only on hlf), so every lane loads unconditionally —
//   no EXEC games, no zero-fill.  All 16 columns of C then hold identical
//   y values; we read column 0 back from lanes 0 and 16:
//     lane 0  : c[r] = y[rowBase + r]      (M = 0..7)
//     lane 16 : c[r] = y[rowBase + 8 + r]  (M = 8..15)
// ---------------------------------------------------------------------------
__device__ __forceinline__ v8f wave_tile_matvec(const __bf16* __restrict__ W,
                                                const __bf16* __restrict__ x,
                                                int rowBase, int K, v8f acc,
                                                unsigned m, unsigned hlf)
{
    const __bf16* wrow = W + (size_t)(rowBase + (int)m) * (size_t)K;
    for (int k0 = 0; k0 < K; k0 += 32) {
        // A: two contiguous 16B chunks per lane, per the ISA A layout
        v8bf alo = *(const v8bf*)(wrow + k0 + hlf * 8);
        v8bf ahi = *(const v8bf*)(wrow + k0 + 16 + hlf * 8);
        v16bf A;
#pragma unroll
        for (int i = 0; i < 8; ++i) { A[i] = alo[i]; A[i + 8] = ahi[i]; }

        // B: broadcast x chunk into every column (unconditional, converged)
        v8bf x0 = *(const v8bf*)(x + k0 + hlf * 8);
        v8bf x1 = *(const v8bf*)(x + k0 + 16 + hlf * 8);
        v16bf B;
#pragma unroll
        for (int i = 0; i < 8; ++i) { B[i] = x0[i]; B[i + 8] = x1[i]; }

        // near-scope prefetch: pull the next A strip into WGP-level cache
        __builtin_prefetch(wrow + k0 + 128, 0, 3);

        acc = __builtin_amdgcn_wmma_f32_16x16x32_bf16(
                  /*neg_a=*/false, A, /*neg_b=*/false, B,
                  /*c_mod=*/(short)0, acc, /*reuse_a=*/false, /*reuse_b=*/false);
    }
    return acc;
}

// ---------------------------------------------------------------------------
// Fused LSTM layer: gates = Wih@x + Whh@h + bih + bhh, then i,f,g,o update.
// Block = 128 threads = 4 waves; wave g owns gate g's 16-row strip for the
// 16 hidden units [blockIdx.x*16, +16). Elementwise update done via LDS.
// ---------------------------------------------------------------------------
__global__ void __launch_bounds__(128)
lstm_layer_kernel(const __bf16* __restrict__ Wih, const __bf16* __restrict__ Whh,
                  const float* __restrict__ bih, const float* __restrict__ bhh,
                  const __bf16* __restrict__ xin, const __bf16* __restrict__ hold,
                  __bf16* __restrict__ hnew, float* __restrict__ c)
{
    __shared__ float gsh[4][16];
    const unsigned tid  = threadIdx.x;
    const unsigned g    = tid >> 5;       // wave id = gate id (i,f,g,o)
    const unsigned lane = tid & 31u;
    const unsigned m    = lane & 15u;
    const unsigned hlf  = lane >> 4;
    const int u0        = (int)blockIdx.x * 16;
    const int rowBase   = (int)g * HDIM + u0;

    v8f acc = {};
    acc = wave_tile_matvec(Wih, xin,  rowBase, HDIM, acc, m, hlf);
    acc = wave_tile_matvec(Whh, hold, rowBase, HDIM, acc, m, hlf);

    if (m == 0) {
#pragma unroll
        for (int r = 0; r < 8; ++r) {
            int u   = (int)hlf * 8 + r;
            int row = rowBase + u;
            gsh[g][u] = acc[r] + bih[row] + bhh[row];
        }
    }
    __syncthreads();

    if (tid < 16) {
        int   U  = u0 + (int)tid;
        float iv = 1.f / (1.f + __expf(-gsh[0][tid]));
        float fv = 1.f / (1.f + __expf(-gsh[1][tid]));
        float gv = tanhf(gsh[2][tid]);
        float ov = 1.f / (1.f + __expf(-gsh[3][tid]));
        float cn = fv * c[U] + iv * gv;
        float hn = ov * tanhf(cn);
        c[U]    = cn;
        hnew[U] = (__bf16)hn;
    }
}

// ---------------------------------------------------------------------------
// logits = 2.5 * tanh((w_pred @ h + b_pred) / 5)
// 256 threads = 8 waves/block, 16 rows per wave -> 250 blocks cover 32000.
// ---------------------------------------------------------------------------
__global__ void __launch_bounds__(256)
logits_kernel(const __bf16* __restrict__ Wp, const float* __restrict__ bp,
              const __bf16* __restrict__ h, float* __restrict__ logits)
{
    const unsigned tid  = threadIdx.x;
    const unsigned w    = tid >> 5;
    const unsigned lane = tid & 31u;
    const unsigned m    = lane & 15u;
    const unsigned hlf  = lane >> 4;
    const int rowBase   = ((int)blockIdx.x * 8 + (int)w) * 16;

    v8f acc = {};
    acc = wave_tile_matvec(Wp, h, rowBase, HDIM, acc, m, hlf);

    if (m == 0) {
#pragma unroll
        for (int r = 0; r < 8; ++r) {
            int row = rowBase + (int)hlf * 8 + r;
            logits[row] = 2.5f * tanhf((acc[r] + bp[row]) * 0.2f);
        }
    }
}

// ---------------------------------------------------------------------------
// Threefry2x32 (JAX default PRNG): 20 rounds, 6 key injections.
// ---------------------------------------------------------------------------
__device__ __forceinline__ unsigned rotl32(unsigned x, int r)
{ return (x << r) | (x >> (32 - r)); }

__device__ __forceinline__ void threefry2x32(unsigned k0, unsigned k1,
                                             unsigned c0, unsigned c1,
                                             unsigned& o0, unsigned& o1)
{
    const unsigned ks2 = 0x1BD11BDAu ^ k0 ^ k1;
    unsigned x0 = c0 + k0, x1 = c1 + k1;
    const int ra[4] = {13, 15, 26, 6};
    const int rb[4] = {17, 29, 16, 24};
#pragma unroll
    for (int i = 0; i < 4; ++i) { x0 += x1; x1 = rotl32(x1, ra[i]); x1 ^= x0; }
    x0 += k1;  x1 += ks2 + 1u;
#pragma unroll
    for (int i = 0; i < 4; ++i) { x0 += x1; x1 = rotl32(x1, rb[i]); x1 ^= x0; }
    x0 += ks2; x1 += k0 + 2u;
#pragma unroll
    for (int i = 0; i < 4; ++i) { x0 += x1; x1 = rotl32(x1, ra[i]); x1 ^= x0; }
    x0 += k0;  x1 += k1 + 3u;
#pragma unroll
    for (int i = 0; i < 4; ++i) { x0 += x1; x1 = rotl32(x1, rb[i]); x1 ^= x0; }
    x0 += k1;  x1 += ks2 + 4u;
#pragma unroll
    for (int i = 0; i < 4; ++i) { x0 += x1; x1 = rotl32(x1, ra[i]); x1 ^= x0; }
    o0 = x0 + ks2; o1 = x1 + k0 + 5u;
}

// ---------------------------------------------------------------------------
// Fused sampler: log-softmax stats + Gumbel-argmax categorical + entropy +
// logp accumulation + embedding gather for the next step's input.
// Single block, 1024 threads. Also advances the PRNG key (split semantics).
// ---------------------------------------------------------------------------
__global__ void __launch_bounds__(1024)
sample_kernel(const float* __restrict__ logits, const float* __restrict__ embd,
              __bf16* __restrict__ xout, unsigned* __restrict__ key,
              float* __restrict__ accum, float* __restrict__ out, int step)
{
    __shared__ float red[1024];
    __shared__ int   redi[1024];
    __shared__ int   sidx;
    const int tid = threadIdx.x;

    // key, sub = jax.random.split(key):
    //   (a0,b0)=tf(key,(0,2)); (a1,b1)=tf(key,(1,3)); new=(a0,a1); sub=(b0,b1)
    const unsigned k0 = key[0], k1 = key[1];
    unsigned a0, b0, a1, b1;
    threefry2x32(k0, k1, 0u, 2u, a0, b0);
    threefry2x32(k0, k1, 1u, 3u, a1, b1);
    const unsigned s0 = b0, s1 = b1;

    // pass 1: global max
    float mloc = -1e30f;
    for (int j = tid; j < NOPS; j += 1024) mloc = fmaxf(mloc, logits[j]);
    red[tid] = mloc; __syncthreads();
    for (int off = 512; off > 0; off >>= 1) {
        if (tid < off) red[tid] = fmaxf(red[tid], red[tid + off]);
        __syncthreads();
    }
    const float mx = red[0]; __syncthreads();

    // pass 2: sum exp, weighted sum, Gumbel argmax
    float sloc = 0.f, wloc = 0.f, best = -1e30f;
    int bidx = 0;
    for (int j = tid; j < NOPS; j += 1024) {
        float l = logits[j];
        float e = __expf(l - mx);
        sloc += e; wloc += l * e;
        unsigned o0, o1, bits;
        if (j < NOPS / 2) { threefry2x32(s0, s1, (unsigned)j, (unsigned)(j + NOPS / 2), o0, o1); bits = o0; }
        else              { threefry2x32(s0, s1, (unsigned)(j - NOPS / 2), (unsigned)j, o0, o1); bits = o1; }
        unsigned fb = (bits >> 9) | 0x3F800000u;       // [1,2)
        float u = __uint_as_float(fb) - 1.0f;          // [0,1)
        u = fmaxf(u, 1.1754944e-38f);
        float cand = l - __logf(-__logf(u));
        if (cand > best) { best = cand; bidx = j; }
    }
    red[tid] = sloc; __syncthreads();
    for (int off = 512; off > 0; off >>= 1) {
        if (tid < off) red[tid] += red[tid + off];
        __syncthreads();
    }
    const float ssum = red[0]; __syncthreads();

    red[tid] = wloc; __syncthreads();
    for (int off = 512; off > 0; off >>= 1) {
        if (tid < off) red[tid] += red[tid + off];
        __syncthreads();
    }
    const float wsum = red[0]; __syncthreads();

    red[tid] = best; redi[tid] = bidx; __syncthreads();
    for (int off = 512; off > 0; off >>= 1) {
        if (tid < off) {
            float a = red[tid], b = red[tid + off];
            if (b > a || (b == a && redi[tid + off] < redi[tid])) {
                red[tid] = b; redi[tid] = redi[tid + off];
            }
        }
        __syncthreads();
    }

    if (tid == 0) {
        int   idx = redi[0];
        sidx = idx;
        float lse = mx + __logf(ssum);
        float lp  = logits[idx] - lse;
        float ent = lse - wsum / ssum;
        float lps = accum[0] + lp;  accum[0] = lps;
        float ens = accum[1] + ent; accum[1] = ens;
        out[0] = lps;
        out[1] = ens;
        out[2 + step] = (float)idx;
        key[0] = a0; key[1] = a1;
    }
    __syncthreads();
    const int idx = sidx;
    xout[tid] = (__bf16)embd[(size_t)idx * HDIM + tid];   // next input
}

// ---------------------------------------------------------------------------
// One-time-per-launch prep (runs every call -> deterministic replay)
// ---------------------------------------------------------------------------
__global__ void cvt_f32_bf16(const float* __restrict__ s, __bf16* __restrict__ d, long n)
{
    long i      = (long)blockIdx.x * blockDim.x + threadIdx.x;
    long stride = (long)gridDim.x * blockDim.x;
    for (; i < n; i += stride) d[i] = (__bf16)s[i];
}

__global__ void __launch_bounds__(1024)
init_kernel(const float* __restrict__ inp, __bf16* xb, __bf16* hA, __bf16* hB,
            float* c, unsigned* key, float* accum)
{
    const int t = threadIdx.x;                 // 1024 threads
    xb[t] = (__bf16)inp[t];
    hA[t] = (__bf16)0.f; hA[t + HDIM] = (__bf16)0.f;
    hB[t] = (__bf16)0.f; hB[t + HDIM] = (__bf16)0.f;
    c[t]  = 0.f;         c[t + HDIM]  = 0.f;
    if (t == 0) { key[0] = 0u; key[1] = 42u; accum[0] = 0.f; accum[1] = 0.f; }
}

// ---------------------------------------------------------------------------
extern "C" void kernel_launch(void* const* d_in, const int* in_sizes, int n_in,
                              void* d_out, int out_size, void* d_ws, size_t ws_size,
                              hipStream_t stream)
{
    (void)in_sizes; (void)n_in; (void)out_size;

    const float* input_vars = (const float*)d_in[0];
    const float* w_ih   = (const float*)d_in[1];   // (2, 4096, 1024)
    const float* w_hh   = (const float*)d_in[2];   // (2, 4096, 1024)
    const float* b_ih   = (const float*)d_in[3];   // (2, 4096)
    const float* b_hh   = (const float*)d_in[4];   // (2, 4096)
    const float* embd   = (const float*)d_in[5];   // (32000, 1024)
    const float* w_pred = (const float*)d_in[6];   // (32000, 1024)
    const float* b_pred = (const float*)d_in[7];   // (32000,)

    // ---- workspace carve-out (256B aligned regions) ----
    char*  ws  = (char*)d_ws;
    size_t cur = 0;
    auto carve = [&](size_t bytes) -> char* {
        char* p = ws + cur;
        cur = (cur + bytes + 255u) & ~(size_t)255u;
        return p;
    };
    const size_t nW   = (size_t)NLSTM * GATES * HDIM;   // 8,388,608
    const size_t nWp  = (size_t)NOPS * HDIM;            // 32,768,000

    __bf16*   wih_b = (__bf16*)carve(nW  * 2);
    __bf16*   whh_b = (__bf16*)carve(nW  * 2);
    __bf16*   wp_b  = (__bf16*)carve(nWp * 2);
    __bf16*   x_b   = (__bf16*)carve(HDIM * 2);
    __bf16*   hA    = (__bf16*)carve((size_t)NLSTM * HDIM * 2);
    __bf16*   hB    = (__bf16*)carve((size_t)NLSTM * HDIM * 2);
    float*    cbuf  = (float*) carve((size_t)NLSTM * HDIM * 4);
    float*    lgts  = (float*) carve((size_t)NOPS * 4);
    unsigned* key   = (unsigned*)carve(8);
    float*    accum = (float*) carve(8);
    if (cur > ws_size) return;   // workspace too small; nothing safe to do

    // ---- phase 0: bf16 weight images (makes all matmul weights L2-resident) ----
    cvt_f32_bf16<<<4096, 256, 0, stream>>>(w_ih,   wih_b, (long)nW);
    cvt_f32_bf16<<<4096, 256, 0, stream>>>(w_hh,   whh_b, (long)nW);
    cvt_f32_bf16<<<8192, 256, 0, stream>>>(w_pred, wp_b,  (long)nWp);
    init_kernel<<<1, 1024, 0, stream>>>(input_vars, x_b, hA, hB, cbuf, key, accum);

    // ---- 128 sequential controller steps ----
    float* out = (float*)d_out;
    for (int s = 0; s < NLAYER; ++s) {
        __bf16* hold = (s & 1) ? hB : hA;     // read-side hidden state
        __bf16* hnew = (s & 1) ? hA : hB;     // write-side hidden state

        // LSTM layer 0: input = x_b
        lstm_layer_kernel<<<64, 128, 0, stream>>>(
            wih_b, whh_b, b_ih, b_hh,
            x_b, hold, hnew, cbuf);

        // LSTM layer 1: input = layer-0's fresh hidden state
        lstm_layer_kernel<<<64, 128, 0, stream>>>(
            wih_b + (size_t)GATES * HDIM, whh_b + (size_t)GATES * HDIM,
            b_ih + GATES, b_hh + GATES,
            hnew, hold + HDIM, hnew + HDIM, cbuf + HDIM);

        // prediction head
        logits_kernel<<<250, 256, 0, stream>>>(wp_b, b_pred, hnew + HDIM, lgts);

        // softmax stats + categorical sample + entropy + embedding gather
        sample_kernel<<<1, 1024, 0, stream>>>(lgts, embd, x_b, key, accum, out, s);
    }
}
